// QMixtralDecoderLayer_59554016526376
// MI455X (gfx1250) — matrix operand order repaired and checked by reading, not compiled, not run on hardware
//
#include <hip/hip_runtime.h>
#include <hip/hip_bf16.h>
#include <cstdint>

typedef __attribute__((ext_vector_type(16))) _Float16 v16h;
typedef __attribute__((ext_vector_type(8)))  _Float16 v8h;
typedef __attribute__((ext_vector_type(8)))  float    v8f;

static constexpr int kB   = 2;
static constexpr int kS   = 2048;
static constexpr int kH   = 2048;
static constexpr int kNH  = 16;
static constexpr int kNKV = 4;
static constexpr int kHD  = 128;
static constexpr int kFFN = 4096;
static constexpr int kNE  = 8;
static constexpr int kT   = kB * kS;       // 4096 tokens
static constexpr int kNA  = kT * 2;        // top-2 assignments

// ---------------------------------------------------------------------------
// gfx1250 async global->LDS copy (ASYNCcnt-tracked), with sync fallback
// Probe result: param 0 is 'int __attribute__((vector_size(16))) __device__ *'
// (global AS1 pointer to int32x4). Use exactly that type, non-const.
// ---------------------------------------------------------------------------
#if defined(__has_builtin)
#if __has_builtin(__builtin_amdgcn_global_load_async_to_lds_b128)
#define HAVE_ASYNC_LDS 1
#endif
#endif
#ifndef HAVE_ASYNC_LDS
#define HAVE_ASYNC_LDS 0
#endif

#if HAVE_ASYNC_LDS
typedef int i32x4 __attribute__((vector_size(16)));
typedef __attribute__((address_space(1))) i32x4 as1_i32x4;
typedef __attribute__((address_space(3))) i32x4 as3_i32x4;
__device__ __forceinline__ void async_copy16(const _Float16* g, _Float16* l) {
  __builtin_amdgcn_global_load_async_to_lds_b128(
      (as1_i32x4*)(uintptr_t)g, (as3_i32x4*)(uint32_t)(uintptr_t)l, 0, 0);
}
#if __has_builtin(__builtin_amdgcn_s_wait_asynccnt)
#define WAIT_ASYNC() __builtin_amdgcn_s_wait_asynccnt(0)
#else
#define WAIT_ASYNC() asm volatile("s_wait_asynccnt 0" ::: "memory")
#endif
#else
__device__ __forceinline__ void async_copy16(const _Float16* g, _Float16* l) {
  *(v8h*)l = *(const v8h*)g;   // global_load_b128 + ds_store_b128
}
#define WAIT_ASYNC() ((void)0)
#endif

// ---------------------------------------------------------------------------
// WMMA helpers (v_wmma_f32_16x16x32_f16)
// ---------------------------------------------------------------------------
__device__ __forceinline__ v8f wmma16(v16h a, v16h b, v8f c) {
  return __builtin_amdgcn_wmma_f32_16x16x32_f16(false, a, false, b,
                                                (short)0, c, false, false);
}

// A fragment: 16x32 f16 tile from row-major [.., lda] storage (global or LDS).
// lane: m = lane&15, half = lane>>4; chunks at k0+half*8 and k0+16+half*8.
__device__ __forceinline__ v16h load_frag_a(const _Float16* base, int lda,
                                            int row0, int k0) {
  int lane = threadIdx.x & 31;
  int m = lane & 15, hf = lane >> 4;
  const _Float16* p = base + (size_t)(row0 + m) * lda + k0 + hf * 8;
  v8h lo = *(const v8h*)(p);
  v8h hi = *(const v8h*)(p + 16);
  v16h r;
#pragma unroll
  for (int i = 0; i < 8; ++i) { r[i] = lo[i]; r[i + 8] = hi[i]; }
  return r;
}

// B fragment: 32x16 tile from W[N][K] row-major (K contiguous per output col).
__device__ __forceinline__ v16h load_frag_b(const _Float16* base, int ldb,
                                            int n0, int k0) {
  int lane = threadIdx.x & 31;
  int n = lane & 15, hf = lane >> 4;
  const _Float16* p = base + (size_t)(n0 + n) * ldb + k0 + hf * 8;
  v8h lo = *(const v8h*)(p);
  v8h hi = *(const v8h*)(p + 16);
  v16h r;
#pragma unroll
  for (int i = 0; i < 8; ++i) { r[i] = lo[i]; r[i + 8] = hi[i]; }
  return r;
}

// C store: VGPR r -> row row0 + r + (lane>=16 ? 8 : 0), col col0 + (lane&15)
__device__ __forceinline__ void store_acc_f32(float* C, int ldc, int row0,
                                              int col0, const v8f& acc,
                                              const float* res) {
  int lane = threadIdx.x & 31;
  int n = lane & 15, hf = lane >> 4;
#pragma unroll
  for (int r = 0; r < 8; ++r) {
    size_t idx = (size_t)(row0 + r + hf * 8) * ldc + col0 + n;
    float v = acc[r];
    if (res) v += res[idx];
    C[idx] = v;
  }
}

// ---------------------------------------------------------------------------
// LDS tile staging (async): A tile 128x32 f16, B tile 64x32 f16, blockDim=256
// ---------------------------------------------------------------------------
__device__ __forceinline__ void stage_a(const _Float16* g, int lda, int row0,
                                        int k0, int rmax, _Float16* s) {
  int t = threadIdx.x;
#pragma unroll
  for (int i = 0; i < 2; ++i) {
    int chunk = t + i * 256;        // 512 chunks of 8 f16
    int r = chunk >> 2;             // 4 chunks per 32-wide row
    int c = (chunk & 3) * 8;
    int row = row0 + r; if (row > rmax) row = rmax;
    async_copy16(g + (size_t)row * lda + k0 + c, s + r * 32 + c);
  }
}

__device__ __forceinline__ void stage_b(const _Float16* g, int ldb, int n0,
                                        int k0, _Float16* s) {
  int t = threadIdx.x;              // 256 chunks of 8 f16
  int r = t >> 2;
  int c = (t & 3) * 8;
  async_copy16(g + (size_t)(n0 + r) * ldb + k0 + c, s + r * 32 + c);
}

// ---------------------------------------------------------------------------
// Elementwise / norm kernels
// ---------------------------------------------------------------------------
__global__ void k_cvt_f16(const float* __restrict__ src,
                          _Float16* __restrict__ dst, size_t n) {
  size_t i = (size_t)blockIdx.x * blockDim.x + threadIdx.x;
  if (i < n) dst[i] = (_Float16)src[i];
}

__global__ void k_rmsnorm(const float* __restrict__ x,
                          const float* __restrict__ w,
                          _Float16* __restrict__ out, int Hdim) {
  int row = blockIdx.x;
  const float* xr = x + (size_t)row * Hdim;
  __shared__ float red[256];
  float ss = 0.f;
  for (int i = threadIdx.x; i < Hdim; i += 256) { float v = xr[i]; ss += v * v; }
  red[threadIdx.x] = ss;
  __syncthreads();
  for (int s = 128; s > 0; s >>= 1) {
    if (threadIdx.x < s) red[threadIdx.x] += red[threadIdx.x + s];
    __syncthreads();
  }
  float inv = rsqrtf(red[0] / (float)Hdim + 1e-5f);
  for (int i = threadIdx.x; i < Hdim; i += 256)
    out[(size_t)row * Hdim + i] = (_Float16)(xr[i] * inv * w[i]);
}

// RoPE + repack [T, nh*HD] f32 -> [B, nh, S, HD] f16 (scale folded in)
__global__ void k_rope_pack(const float* __restrict__ src,
                            const float* __restrict__ cosb,
                            const float* __restrict__ sinb,
                            _Float16* __restrict__ dst, int nh, float scale) {
  size_t i = (size_t)blockIdx.x * blockDim.x + threadIdx.x;
  size_t total = (size_t)kT * nh * kHD;
  if (i >= total) return;
  int d = (int)(i % kHD);
  size_t tmp = i / kHD;
  int hh = (int)(tmp % nh);
  size_t t = tmp / nh;
  int s = (int)(t % kS);
  int b = (int)(t / kS);
  size_t rowoff = t * ((size_t)nh * kHD) + (size_t)hh * kHD;
  float xv = src[rowoff + d];
  float xo = (d < kHD / 2) ? -src[rowoff + d + kHD / 2]
                           :  src[rowoff + d - kHD / 2];
  float c = cosb[(size_t)s * kHD + d];
  float sn = sinb[(size_t)s * kHD + d];
  float r = (xv * c + xo * sn) * scale;
  dst[(((size_t)b * nh + hh) * kS + s) * kHD + d] = (_Float16)r;
}

// V f32 [T, NKV*HD] -> Vt f16 [B, NKV, HD, S]
__global__ void k_pack_vt(const float* __restrict__ v,
                          _Float16* __restrict__ vt) {
  size_t i = (size_t)blockIdx.x * blockDim.x + threadIdx.x;
  size_t total = (size_t)kT * kNKV * kHD;
  if (i >= total) return;
  int d = (int)(i % kHD);
  size_t tmp = i / kHD;
  int hh = (int)(tmp % kNKV);
  size_t t = tmp / kNKV;
  int s = (int)(t % kS);
  int b = (int)(t / kS);
  float val = v[t * ((size_t)kNKV * kHD) + (size_t)hh * kHD + d];
  vt[((((size_t)b * kNKV + hh) * kHD + d) * kS) + s] = (_Float16)val;
}

// ---------------------------------------------------------------------------
// LDS-staged WMMA GEMM: C[M,N](f32, +opt residual) = A16[M,K] x W16[N,K]^T
// block = 256 threads = 8 waves as 4(M)x2(N); macro-tile 128x64; double-buffer
// async global->LDS staging, K-step 32.
// ---------------------------------------------------------------------------
__global__ __launch_bounds__(256)
void k_gemm_lds(const _Float16* __restrict__ A, const _Float16* __restrict__ W,
                float* __restrict__ C, const float* __restrict__ res,
                int M, int N, int K) {
  __shared__ _Float16 sA[2][128 * 32];
  __shared__ _Float16 sB[2][64 * 32];
  int mt_blk = blockIdx.y * 128;
  int nt_blk = blockIdx.x * 64;
  int wave = threadIdx.x >> 5;
  int lr = (wave >> 1) * 32;       // wave row within macro-tile
  int lc = (wave & 1) * 32;        // wave col within macro-tile
  int rmax = M - 1;
  v8f zero = {};
  v8f acc[2][2];
#pragma unroll
  for (int i = 0; i < 2; ++i)
#pragma unroll
    for (int j = 0; j < 2; ++j) acc[i][j] = zero;

  stage_a(A, K, mt_blk, 0, rmax, sA[0]);
  stage_b(W, K, nt_blk, 0, sB[0]);
  int buf = 0;
  for (int kk = 0; kk < K; kk += 32) {
    WAIT_ASYNC();
    __syncthreads();
    if (kk + 32 < K) {
      stage_a(A, K, mt_blk, kk + 32, rmax, sA[buf ^ 1]);
      stage_b(W, K, nt_blk, kk + 32, sB[buf ^ 1]);
    }
    const _Float16* la = sA[buf];
    const _Float16* lb = sB[buf];
    v16h a0 = load_frag_a(la, 32, lr, 0);
    v16h a1 = load_frag_a(la, 32, lr + 16, 0);
    v16h b0 = load_frag_b(lb, 32, lc, 0);
    v16h b1 = load_frag_b(lb, 32, lc + 16, 0);
    acc[0][0] = wmma16(a0, b0, acc[0][0]);
    acc[0][1] = wmma16(a0, b1, acc[0][1]);
    acc[1][0] = wmma16(a1, b0, acc[1][0]);
    acc[1][1] = wmma16(a1, b1, acc[1][1]);
    buf ^= 1;
  }
  int mw = mt_blk + lr, nw = nt_blk + lc;
  store_acc_f32(C, N, mw,      nw,      acc[0][0], res);
  store_acc_f32(C, N, mw,      nw + 16, acc[0][1], res);
  store_acc_f32(C, N, mw + 16, nw,      acc[1][0], res);
  store_acc_f32(C, N, mw + 16, nw + 16, acc[1][1], res);
}

// ---------------------------------------------------------------------------
// Flash attention (one wave per 16-query tile, causal, online softmax).
// Scores computed transposed (K·Q^T) so each lane's score VGPRs are exactly
// the A-fragment chunks of P for the P·V WMMA — no cross-lane repack.
// q16 pre-scaled by HD^-0.5 * log2(e); exp2 inner loop.
// ---------------------------------------------------------------------------
__global__ void k_flash(const _Float16* __restrict__ q16,
                        const _Float16* __restrict__ k16,
                        const _Float16* __restrict__ vt16,
                        _Float16* __restrict__ attn_out) {
  int qt = blockIdx.x;            // S/16 tiles
  int h  = blockIdx.y;            // NH
  int b  = blockIdx.z;            // B
  int kvh = h / (kNH / kNKV);
  const _Float16* Q  = q16  + ((size_t)b * kNH  + h)   * kS * kHD;   // [S][HD]
  const _Float16* Kh = k16  + ((size_t)b * kNKV + kvh) * kS * kHD;   // [S][HD]
  const _Float16* Vt = vt16 + ((size_t)b * kNKV + kvh) * kHD * (size_t)kS; // [HD][S]

  int lane = threadIdx.x & 31;
  int nq = lane & 15, half = lane >> 4;
  int q0 = qt * 16;
  int qg = q0 + nq;               // this lane's query row

  v16h qfrag[4];
#pragma unroll
  for (int c = 0; c < 4; ++c) qfrag[c] = load_frag_b(Q, kHD, q0, c * 32);

  v8f zero = {};
  v8f o[8];
#pragma unroll
  for (int d = 0; d < 8; ++d) o[d] = zero;
  float m_run = -1e30f, l_run = 0.f;

  int kend = q0 + 16;             // need keys <= q0+15
  for (int kb = 0; kb < kend; kb += 32) {
    v8f s0 = zero, s1 = zero;
#pragma unroll
    for (int c = 0; c < 4; ++c) {
      v16h ka0 = load_frag_a(Kh, kHD, kb,      c * 32);
      v16h ka1 = load_frag_a(Kh, kHD, kb + 16, c * 32);
      s0 = wmma16(ka0, qfrag[c], s0);   // M=key(kb..kb+15), N=query
      s1 = wmma16(ka1, qfrag[c], s1);   // M=key(kb+16..kb+31)
    }
    float tmax = -1e30f;
#pragma unroll
    for (int r = 0; r < 8; ++r) {
      int key0 = kb + r + 8 * half;
      int key1 = key0 + 16;
      if (key0 > qg) s0[r] = -1e30f;
      if (key1 > qg) s1[r] = -1e30f;
      tmax = fmaxf(tmax, fmaxf(s0[r], s1[r]));
    }
    tmax = fmaxf(tmax, __shfl_xor(tmax, 16, 32));
    float mnew = fmaxf(m_run, tmax);
    float alpha = exp2f(m_run - mnew);
    float lsum = 0.f;
    v16h p;                          // A-fragment of P (16 queries x 32 keys)
#pragma unroll
    for (int r = 0; r < 8; ++r) {
      float p0 = exp2f(s0[r] - mnew);
      float p1 = exp2f(s1[r] - mnew);
      lsum += p0 + p1;
      p[r]     = (_Float16)p0;
      p[r + 8] = (_Float16)p1;
    }
    lsum += __shfl_xor(lsum, 16, 32);
    l_run = l_run * alpha + lsum;
    m_run = mnew;
    float af[8];
#pragma unroll
    for (int r = 0; r < 8; ++r) af[r] = __shfl(alpha, r + 8 * half, 32);
#pragma unroll
    for (int d = 0; d < 8; ++d) {
#pragma unroll
      for (int r = 0; r < 8; ++r) o[d][r] *= af[r];
      v16h vb = load_frag_b(Vt, kS, d * 16, kb);   // B: n=dim, K=keys
      o[d] = wmma16(p, vb, o[d]);
    }
  }
  float linv = 1.f / l_run;
  float lf[8];
#pragma unroll
  for (int r = 0; r < 8; ++r) lf[r] = __shfl(linv, r + 8 * half, 32);
#pragma unroll
  for (int d = 0; d < 8; ++d) {
#pragma unroll
    for (int r = 0; r < 8; ++r) {
      int srow = q0 + r + 8 * half;
      size_t idx = ((size_t)b * kS + srow) * (kNH * kHD) + h * kHD + d * 16 + nq;
      attn_out[idx] = (_Float16)(o[d][r] * lf[r]);
    }
  }
}

// ---------------------------------------------------------------------------
// MoE routing: logits, softmax, top-2, normalized gates
// ---------------------------------------------------------------------------
__global__ void k_router(const _Float16* __restrict__ x2,
                         const float* __restrict__ rw,
                         int* __restrict__ topi, float* __restrict__ topw) {
  int t = blockIdx.x;
  int tid = threadIdx.x;              // 64 threads
  __shared__ float part[kNE][64];
  float acc[kNE] = {0, 0, 0, 0, 0, 0, 0, 0};
  const _Float16* xr = x2 + (size_t)t * kH;
  for (int i = tid; i < kH; i += 64) {
    float xv = (float)xr[i];
#pragma unroll
    for (int e = 0; e < kNE; ++e) acc[e] += xv * rw[(size_t)e * kH + i];
  }
#pragma unroll
  for (int e = 0; e < kNE; ++e) part[e][tid] = acc[e];
  __syncthreads();
  if (tid < kNE) {
    float s = 0.f;
    for (int j = 0; j < 64; ++j) s += part[tid][j];
    part[tid][0] = s;
  }
  __syncthreads();
  if (tid == 0) {
    float mx = -1e30f;
    for (int e = 0; e < kNE; ++e) mx = fmaxf(mx, part[e][0]);
    float pr[kNE], se = 0.f;
    for (int e = 0; e < kNE; ++e) { pr[e] = expf(part[e][0] - mx); se += pr[e]; }
    for (int e = 0; e < kNE; ++e) pr[e] /= se;
    int i0 = 0;
    for (int e = 1; e < kNE; ++e) if (pr[e] > pr[i0]) i0 = e;
    int i1 = (i0 == 0) ? 1 : 0;
    for (int e = 0; e < kNE; ++e)
      if (e != i0 && pr[e] > pr[i1]) i1 = e;
    float sw = pr[i0] + pr[i1];
    topi[t * 2] = i0; topi[t * 2 + 1] = i1;
    topw[t * 2] = pr[i0] / sw; topw[t * 2 + 1] = pr[i1] / sw;
  }
}

__global__ void k_zero_i32(int* p, int n) {
  int i = blockIdx.x * blockDim.x + threadIdx.x;
  if (i < n) p[i] = 0;
}

__global__ void k_count(const int* __restrict__ topi, int* __restrict__ counts) {
  int i = blockIdx.x * blockDim.x + threadIdx.x;
  if (i < kNA) atomicAdd(&counts[topi[i]], 1);
}

__global__ void k_scan(const int* __restrict__ counts,
                       int* __restrict__ segstart, int* __restrict__ running) {
  if (threadIdx.x == 0 && blockIdx.x == 0) {
    int a = 0;
    for (int e = 0; e < kNE; ++e) { segstart[e] = a; running[e] = a; a += counts[e]; }
    segstart[kNE] = a;
  }
}

__global__ void k_scatter(const int* __restrict__ topi,
                          const float* __restrict__ topw,
                          int* __restrict__ running, int* __restrict__ rows,
                          float* __restrict__ gaterows,
                          int* __restrict__ tokslot) {
  int i = blockIdx.x * blockDim.x + threadIdx.x;
  if (i >= kNA) return;
  int e = topi[i];
  int pos = atomicAdd(&running[e], 1);
  rows[pos] = i >> 1;
  gaterows[pos] = topw[i];
  tokslot[i] = pos;
}

__global__ void k_gather(const _Float16* __restrict__ x2,
                         const int* __restrict__ rows,
                         _Float16* __restrict__ xg) {
  size_t i = (size_t)blockIdx.x * blockDim.x + threadIdx.x;
  if (i >= (size_t)kNA * kH) return;
  int col = (int)(i % kH);
  int p = (int)(i / kH);
  xg[(size_t)p * kH + col] = x2[(size_t)rows[p] * kH + col];
}

// ---------------------------------------------------------------------------
// MoE MLP1 (LDS-staged, dual-B): h = silu(xg·w1^T) * (xg·w3^T) * gate
// macro-tile 128(rows) x 64(FFN cols); async double-buffered staging.
// ---------------------------------------------------------------------------
__global__ __launch_bounds__(256)
void k_moe_mlp1(const _Float16* __restrict__ xg,
                const _Float16* __restrict__ w1,
                const _Float16* __restrict__ w3,
                const float* __restrict__ gates,
                const int* __restrict__ segstart,
                _Float16* __restrict__ hbuf) {
  __shared__ _Float16 sA[2][128 * 32];
  __shared__ _Float16 sB1[2][64 * 32];
  __shared__ _Float16 sB3[2][64 * 32];
  int e = blockIdx.z;
  int s0 = segstart[e], s1 = segstart[e + 1];
  int mt_blk = s0 + blockIdx.y * 128;
  if (mt_blk >= s1) return;                 // block-uniform
  int nt_blk = blockIdx.x * 64;
  const _Float16* W1 = w1 + (size_t)e * kFFN * kH;
  const _Float16* W3 = w3 + (size_t)e * kFFN * kH;
  int wave = threadIdx.x >> 5;
  int lr = (wave >> 1) * 32;
  int lc = (wave & 1) * 32;
  int rmax = s1 - 1;
  v8f zero = {};
  v8f ga[2][2], gb[2][2];
#pragma unroll
  for (int i = 0; i < 2; ++i)
#pragma unroll
    for (int j = 0; j < 2; ++j) { ga[i][j] = zero; gb[i][j] = zero; }

  stage_a(xg, kH, mt_blk, 0, rmax, sA[0]);
  stage_b(W1, kH, nt_blk, 0, sB1[0]);
  stage_b(W3, kH, nt_blk, 0, sB3[0]);
  int buf = 0;
  for (int kk = 0; kk < kH; kk += 32) {
    WAIT_ASYNC();
    __syncthreads();
    if (kk + 32 < kH) {
      stage_a(xg, kH, mt_blk, kk + 32, rmax, sA[buf ^ 1]);
      stage_b(W1, kH, nt_blk, kk + 32, sB1[buf ^ 1]);
      stage_b(W3, kH, nt_blk, kk + 32, sB3[buf ^ 1]);
    }
    v16h x0 = load_frag_a(sA[buf], 32, lr, 0);
    v16h x1 = load_frag_a(sA[buf], 32, lr + 16, 0);
    v16h u0 = load_frag_b(sB1[buf], 32, lc, 0);
    v16h u1 = load_frag_b(sB1[buf], 32, lc + 16, 0);
    v16h t0 = load_frag_b(sB3[buf], 32, lc, 0);
    v16h t1 = load_frag_b(sB3[buf], 32, lc + 16, 0);
    ga[0][0] = wmma16(x0, u0, ga[0][0]); ga[0][1] = wmma16(x0, u1, ga[0][1]);
    ga[1][0] = wmma16(x1, u0, ga[1][0]); ga[1][1] = wmma16(x1, u1, ga[1][1]);
    gb[0][0] = wmma16(x0, t0, gb[0][0]); gb[0][1] = wmma16(x0, t1, gb[0][1]);
    gb[1][0] = wmma16(x1, t0, gb[1][0]); gb[1][1] = wmma16(x1, t1, gb[1][1]);
    buf ^= 1;
  }
  int lane = threadIdx.x & 31;
  int n = lane & 15, hf = lane >> 4;
  int mw = mt_blk + lr, nw = nt_blk + lc;
#pragma unroll
  for (int i = 0; i < 2; ++i) {
#pragma unroll
    for (int r = 0; r < 8; ++r) {
      int row = mw + i * 16 + r + hf * 8;
      if (row < s1) {
        float g = gates[row];
#pragma unroll
        for (int j = 0; j < 2; ++j) {
          float av = ga[i][j][r], bv = gb[i][j][r];
          float hv = (av / (1.f + expf(-av))) * bv * g;
          hbuf[(size_t)row * kFFN + nw + j * 16 + n] = (_Float16)hv;
        }
      }
    }
  }
}

// ---------------------------------------------------------------------------
// MoE MLP2 (LDS-staged): y = h·w2^T (per expert segment), f32 per-slot rows
// ---------------------------------------------------------------------------
__global__ __launch_bounds__(256)
void k_moe_mlp2(const _Float16* __restrict__ hbuf,
                const _Float16* __restrict__ w2,
                const int* __restrict__ segstart,
                float* __restrict__ y) {
  __shared__ _Float16 sA[2][128 * 32];
  __shared__ _Float16 sB[2][64 * 32];
  int e = blockIdx.z;
  int s0 = segstart[e], s1 = segstart[e + 1];
  int mt_blk = s0 + blockIdx.y * 128;
  if (mt_blk >= s1) return;                 // block-uniform
  int nt_blk = blockIdx.x * 64;
  const _Float16* W2 = w2 + (size_t)e * kH * kFFN;   // [H][FFN]
  int wave = threadIdx.x >> 5;
  int lr = (wave >> 1) * 32;
  int lc = (wave & 1) * 32;
  int rmax = s1 - 1;
  v8f zero = {};
  v8f acc[2][2];
#pragma unroll
  for (int i = 0; i < 2; ++i)
#pragma unroll
    for (int j = 0; j < 2; ++j) acc[i][j] = zero;

  stage_a(hbuf, kFFN, mt_blk, 0, rmax, sA[0]);
  stage_b(W2, kFFN, nt_blk, 0, sB[0]);
  int buf = 0;
  for (int kk = 0; kk < kFFN; kk += 32) {
    WAIT_ASYNC();
    __syncthreads();
    if (kk + 32 < kFFN) {
      stage_a(hbuf, kFFN, mt_blk, kk + 32, rmax, sA[buf ^ 1]);
      stage_b(W2, kFFN, nt_blk, kk + 32, sB[buf ^ 1]);
    }
    v16h a0 = load_frag_a(sA[buf], 32, lr, 0);
    v16h a1 = load_frag_a(sA[buf], 32, lr + 16, 0);
    v16h b0 = load_frag_b(sB[buf], 32, lc, 0);
    v16h b1 = load_frag_b(sB[buf], 32, lc + 16, 0);
    acc[0][0] = wmma16(a0, b0, acc[0][0]); acc[0][1] = wmma16(a0, b1, acc[0][1]);
    acc[1][0] = wmma16(a1, b0, acc[1][0]); acc[1][1] = wmma16(a1, b1, acc[1][1]);
    buf ^= 1;
  }
  int lane = threadIdx.x & 31;
  int n = lane & 15, hf = lane >> 4;
  int mw = mt_blk + lr, nw = nt_blk + lc;
#pragma unroll
  for (int i = 0; i < 2; ++i)
#pragma unroll
    for (int r = 0; r < 8; ++r) {
      int row = mw + i * 16 + r + hf * 8;
      if (row < s1) {
#pragma unroll
        for (int j = 0; j < 2; ++j)
          y[(size_t)row * kH + nw + j * 16 + n] = acc[i][j][r];
      }
    }
}

// out = hidden2 + y[slot0] + y[slot1]
__global__ void k_final(const float* __restrict__ hidden2,
                        const float* __restrict__ y,
                        const int* __restrict__ tokslot,
                        float* __restrict__ out) {
  size_t i = (size_t)blockIdx.x * blockDim.x + threadIdx.x;
  if (i >= (size_t)kT * kH) return;
  int col = (int)(i % kH);
  int t = (int)(i / kH);
  int p0 = tokslot[t * 2], p1 = tokslot[t * 2 + 1];
  out[i] = hidden2[i] + y[(size_t)p0 * kH + col] + y[(size_t)p1 * kH + col];
}

// ---------------------------------------------------------------------------
// Host launcher
// ---------------------------------------------------------------------------
extern "C" void kernel_launch(void* const* d_in, const int* in_sizes, int n_in,
                              void* d_out, int out_size, void* d_ws,
                              size_t ws_size, hipStream_t stream) {
  (void)in_sizes; (void)n_in; (void)out_size; (void)ws_size;
  const float* hidden   = (const float*)d_in[0];
  const float* ln1_w    = (const float*)d_in[1];
  const float* ln2_w    = (const float*)d_in[2];
  const float* wq       = (const float*)d_in[3];
  const float* wk       = (const float*)d_in[4];
  const float* wv       = (const float*)d_in[5];
  const float* wo       = (const float*)d_in[6];
  const float* router_w = (const float*)d_in[7];
  const float* w1       = (const float*)d_in[8];
  const float* w3       = (const float*)d_in[9];
  const float* w2       = (const float*)d_in[10];
  const float* cosb     = (const float*)d_in[11];
  const float* sinb     = (const float*)d_in[12];
  float* out = (float*)d_out;

  char* wsp = (char*)d_ws;
  size_t off = 0;
  auto alloc = [&](size_t bytes) -> void* {
    void* r = (void*)(wsp + off);
    off += (bytes + 255) & ~(size_t)255;
    return r;
  };

  _Float16* wq16  = (_Float16*)alloc((size_t)2048 * 2048 * 2);
  _Float16* wk16  = (_Float16*)alloc((size_t)512 * 2048 * 2);
  _Float16* wv16  = (_Float16*)alloc((size_t)512 * 2048 * 2);
  _Float16* wo16  = (_Float16*)alloc((size_t)2048 * 2048 * 2);
  _Float16* w1_16 = (_Float16*)alloc((size_t)kNE * kFFN * kH * 2);
  _Float16* w3_16 = (_Float16*)alloc((size_t)kNE * kFFN * kH * 2);
  _Float16* w2_16 = (_Float16*)alloc((size_t)kNE * kH * kFFN * 2);
  _Float16* xn16  = (_Float16*)alloc((size_t)kT * kH * 2);
  float*    qf32  = (float*)alloc((size_t)kT * kNH * kHD * 4);
  float*    kf32  = (float*)alloc((size_t)kT * kNKV * kHD * 4);
  float*    vf32  = (float*)alloc((size_t)kT * kNKV * kHD * 4);
  _Float16* q16   = (_Float16*)alloc((size_t)kT * kNH * kHD * 2);
  _Float16* k16   = (_Float16*)alloc((size_t)kT * kNKV * kHD * 2);
  _Float16* vt16  = (_Float16*)alloc((size_t)kT * kNKV * kHD * 2);
  _Float16* attn16 = (_Float16*)alloc((size_t)kT * kNH * kHD * 2);
  float*    hidden2 = (float*)alloc((size_t)kT * kH * 4);
  _Float16* x2_16 = (_Float16*)alloc((size_t)kT * kH * 2);
  int*      topi  = (int*)alloc((size_t)kNA * 4);
  float*    topw  = (float*)alloc((size_t)kNA * 4);
  int*      counts   = (int*)alloc(kNE * 4);
  int*      segstart = (int*)alloc((kNE + 1) * 4);
  int*      running  = (int*)alloc(kNE * 4);
  int*      rows     = (int*)alloc((size_t)kNA * 4);
  float*    gaterows = (float*)alloc((size_t)kNA * 4);
  int*      tokslot  = (int*)alloc((size_t)kNA * 4);
  _Float16* xg16  = (_Float16*)alloc((size_t)kNA * kH * 2);
  _Float16* h16   = (_Float16*)alloc((size_t)kNA * kFFN * 2);
  float*    ybuf  = (float*)alloc((size_t)kNA * kH * 4);

  auto cvt = [&](const float* s, _Float16* d, size_t n) {
    k_cvt_f16<<<(unsigned)((n + 255) / 256), 256, 0, stream>>>(s, d, n);
  };
  cvt(wq, wq16, (size_t)2048 * 2048);
  cvt(wk, wk16, (size_t)512 * 2048);
  cvt(wv, wv16, (size_t)512 * 2048);
  cvt(wo, wo16, (size_t)2048 * 2048);
  cvt(w1, w1_16, (size_t)kNE * kFFN * kH);
  cvt(w3, w3_16, (size_t)kNE * kFFN * kH);
  cvt(w2, w2_16, (size_t)kNE * kH * kFFN);

  // 1) ln1 -> xn16
  k_rmsnorm<<<kT, 256, 0, stream>>>(hidden, ln1_w, xn16, kH);

  // 2) QKV projections (LDS-staged WMMA GEMMs)
  k_gemm_lds<<<dim3(2048 / 64, kT / 128), 256, 0, stream>>>(xn16, wq16, qf32, nullptr, kT, 2048, kH);
  k_gemm_lds<<<dim3(512 / 64,  kT / 128), 256, 0, stream>>>(xn16, wk16, kf32, nullptr, kT, 512, kH);
  k_gemm_lds<<<dim3(512 / 64,  kT / 128), 256, 0, stream>>>(xn16, wv16, vf32, nullptr, kT, 512, kH);

  // 3) RoPE + layout packs (fold softmax scale * log2(e) into Q)
  float qscale = rsqrtf((float)kHD) * 1.44269504088896340736f;
  {
    size_t nq = (size_t)kT * kNH * kHD;
    k_rope_pack<<<(unsigned)((nq + 255) / 256), 256, 0, stream>>>(qf32, cosb, sinb, q16, kNH, qscale);
    size_t nk = (size_t)kT * kNKV * kHD;
    k_rope_pack<<<(unsigned)((nk + 255) / 256), 256, 0, stream>>>(kf32, cosb, sinb, k16, kNKV, 1.0f);
    k_pack_vt<<<(unsigned)((nk + 255) / 256), 256, 0, stream>>>(vf32, vt16);
  }

  // 4) Flash attention
  k_flash<<<dim3(kS / 16, kNH, kB), 32, 0, stream>>>(q16, k16, vt16, attn16);

  // 5) Output projection + residual
  k_gemm_lds<<<dim3(2048 / 64, kT / 128), 256, 0, stream>>>(attn16, wo16, hidden2, hidden, kT, kH, 2048);

  // 6) ln2 -> x2_16
  k_rmsnorm<<<kT, 256, 0, stream>>>(hidden2, ln2_w, x2_16, kH);

  // 7) Router + top-2 gather
  k_router<<<kT, 64, 0, stream>>>(x2_16, router_w, topi, topw);
  k_zero_i32<<<1, 32, 0, stream>>>(counts, kNE);
  k_count<<<(kNA + 255) / 256, 256, 0, stream>>>(topi, counts);
  k_scan<<<1, 32, 0, stream>>>(counts, segstart, running);
  k_scatter<<<(kNA + 255) / 256, 256, 0, stream>>>(topi, topw, running, rows, gaterows, tokslot);
  {
    size_t ng = (size_t)kNA * kH;
    k_gather<<<(unsigned)((ng + 255) / 256), 256, 0, stream>>>(x2_16, rows, xg16);
  }

  // 8) Expert FFNs (segmented, LDS-staged WMMA GEMMs)
  k_moe_mlp1<<<dim3(kFFN / 64, kT / 128, kNE), 256, 0, stream>>>(xg16, w1_16, w3_16, gaterows, segstart, h16);
  k_moe_mlp2<<<dim3(kH / 64, kT / 128, kNE), 256, 0, stream>>>(h16, w2_16, segstart, ybuf);

  // 9) Final residual sum
  {
    size_t nf = (size_t)kT * kH;
    k_final<<<(unsigned)((nf + 255) / 256), 256, 0, stream>>>(hidden2, ybuf, tokslot, out);
  }
}
